// TranslationRNN_8821862826734
// MI455X (gfx1250) — matrix operand order repaired and checked by reading, not compiled, not run on hardware
//
#include <hip/hip_runtime.h>

// ---------------- types ----------------
typedef __attribute__((ext_vector_type(16))) __bf16 v16bf;
typedef __attribute__((ext_vector_type(8)))  float  v8f;

// ---------------- model constants ----------------
constexpr int GB   = 32;     // batch
constexpr int GH   = 1024;   // hidden == embed
constexpr int GK2  = 2048;   // 2H
constexpr int SSRC = 128;
constexpr int STGT = 64;
constexpr int VTGT = 32000;
constexpr int NTILE = GH / 16;        // 64 column tiles for GRU GEMMs
constexpr int KTH   = GH / 32;        // 32 k-tiles per 1024 of K
constexpr int FRAG_ELEMS  = 512;      // 32 lanes * 16 halfs per 32x16 / 16x32 fragment tile
constexpr int TSTEP_FRAGS = KTH * 2;  // 64 fragment tiles per 32x1024 activation panel
constexpr int GRU_BLOCKS  = 16;       // 16 blocks * 8 waves = 128 waves
constexpr int NTB = VTGT / 16;        // 2000 N-tiles in projection

// ---------------- helpers ----------------
__device__ __forceinline__ unsigned short f2bf(float f) {
  union { float f; unsigned u; } v; v.f = f;
  unsigned r = v.u + 0x7FFFu + ((v.u >> 16) & 1u);  // RNE
  return (unsigned short)(r >> 16);
}
__device__ __forceinline__ float sigm(float x) { return 1.0f / (1.0f + __expf(-x)); }

__device__ __forceinline__ v8f wmma_bf16(v16bf a, v16bf b, v8f c) {
  return __builtin_amdgcn_wmma_f32_16x16x32_bf16(false, a, false, b, (short)0, c, false, false);
}

// A-fragment K swizzle (16-bit A 16x32 VGPR layout): k offset held by (lane, j)
__device__ __forceinline__ int kidxA(int lane, int j) {
  return (j & 7) + ((j >> 3) << 4) + ((lane >> 4) << 3);
}

// scatter one fp32 value into a bf16 A-fragment panel (32 rows x 1024 cols)
__device__ __forceinline__ void store_afrag(unsigned short* base, int m, int n, float v) {
  const int kt = n >> 5, ko = n & 31;
  const int lp = (m & 15) | (((ko >> 3) & 1) << 4);
  const int j  = (ko & 7) | ((ko >> 4) << 3);
  const int mt = m >> 4;
  base[((size_t)(kt * 2 + mt) * 32 + lp) * 16 + j] = f2bf(v);
}

// load a fragment from a per-lane base pointer (base + lane*16) at tile offset
__device__ __forceinline__ v16bf ldf(const unsigned short* lane_base, size_t tile) {
  return *(const v16bf*)(lane_base + tile * FRAG_ELEMS);
}

// grid-wide barrier (monotone counter; counter zeroed by init kernel each launch)
__device__ __forceinline__ void gbarrier(unsigned* cnt, unsigned& gen, unsigned nblocks) {
  __syncthreads();
  gen++;
  if (threadIdx.x == 0) {
    __threadfence();
    atomicAdd(cnt, 1u);
    const unsigned target = gen * nblocks;
    volatile unsigned* vc = cnt;
    while (*vc < target) { __builtin_amdgcn_s_sleep(2); }
    __threadfence();
  }
  __syncthreads();
}

// ---------------- init: zero barriers, fp32 h, bf16 h-fragments ----------------
__global__ void init_ws(unsigned* __restrict__ bar, float* __restrict__ hbuf,
                        unsigned* __restrict__ hfrag_u32) {
  const int tid = blockIdx.x * blockDim.x + threadIdx.x;
  const int stride = gridDim.x * blockDim.x;
  if (tid < 64) bar[tid] = 0u;
  for (int i = tid; i < 2 * GB * GH; i += stride) hbuf[i] = 0.0f;
  for (int i = tid; i < 2 * TSTEP_FRAGS * FRAG_ELEMS / 2; i += stride) hfrag_u32[i] = 0u;
}

// ---------------- embedding gather -> bf16 A-fragment sequence ----------------
__global__ void gather_embed_frag(const float* __restrict__ emb, const int* __restrict__ idx,
                                  unsigned short* __restrict__ outf, int S) {
  const int gid = blockIdx.x * blockDim.x + threadIdx.x;  // S * 64 * 32 total
  const int lane = gid & 31;
  const int f = (gid >> 5) & 63;
  const int t = gid >> 11;
  const int mt = f & 1, kt = f >> 1;
  const int b = mt * 16 + (lane & 15);
  const int vrow = idx[b * S + t];
  const float* src = emb + (size_t)vrow * GH + kt * 32;
  __align__(16) unsigned short tmp[16];
#pragma unroll
  for (int j = 0; j < 16; ++j) tmp[j] = f2bf(src[kidxA(lane, j)]);
  uint4* d = (uint4*)(outf + ((size_t)(t * TSTEP_FRAGS + f) * 32 + lane) * 16);
  d[0] = *(const uint4*)&tmp[0];
  d[1] = *(const uint4*)&tmp[8];
}

// ---------------- weight pack: fp32 [K,N] row-major -> bf16 B-fragment tiles ----------------
__global__ void pack_weight_b(const float* __restrict__ src, unsigned short* __restrict__ dst,
                              int K, int N) {
  const int total = (K >> 5) * (N >> 4) * 32;
  const int gid = blockIdx.x * blockDim.x + threadIdx.x;
  if (gid >= total) return;
  const int ln = gid & 31;
  const int tile = gid >> 5;
  const int ntn = N >> 4;
  const int kt = tile / ntn;
  const int nt = tile - kt * ntn;
  const int n = nt * 16 + (ln & 15);
  const int kbase = kt * 32;
  __align__(16) unsigned short tmp[16];
#pragma unroll
  for (int j = 0; j < 16; ++j) tmp[j] = f2bf(src[(size_t)(kbase + kidxA(ln, j)) * N + n]);
  uint4* d = (uint4*)(dst + (size_t)tile * FRAG_ELEMS + (size_t)ln * 16);
  d[0] = *(const uint4*)&tmp[0];
  d[1] = *(const uint4*)&tmp[8];
}

// ---------------- persistent GRU layer kernel (global bf16 fragments, pipelined) ----------------
__global__ __launch_bounds__(256) void gru_layer(
    const unsigned short* __restrict__ xf, unsigned short* __restrict__ yf, int S,
    float* __restrict__ h, unsigned short* __restrict__ hfrag,
    unsigned short* __restrict__ rhfrag,
    const unsigned short* __restrict__ Wz, const unsigned short* __restrict__ Wr,
    const unsigned short* __restrict__ Wh,
    const float* __restrict__ bz, const float* __restrict__ br, const float* __restrict__ bh,
    float* __restrict__ zbuf, float* __restrict__ hxbuf,
    unsigned* __restrict__ bar, unsigned nblocks) {
  const int lane = threadIdx.x & 31;
  const int wgid = blockIdx.x * (blockDim.x >> 5) + (threadIdx.x >> 5);  // 0..127
  unsigned gen = 0;

  const bool isZ = (wgid < 64);              // z tiles 0..63 (also own hx), r tiles 64..127
  const int ntA = isZ ? wgid : wgid - 64;
  const unsigned short* WA = isZ ? Wz : Wr;

  // per-wave lane base pointers; per-kt weight stride = NTILE tiles
  const size_t KS = (size_t)NTILE * FRAG_ELEMS;  // 32768 elements per weight k-row
  const unsigned short* wAL = WA + (size_t)ntA * FRAG_ELEMS + (size_t)lane * 16;
  const unsigned short* wBL = Wh + (size_t)ntA * FRAG_ELEMS + (size_t)lane * 16;  // x-rows of Wh
  const unsigned short* wCL = Wh + (size_t)(KTH * NTILE + ntA) * FRAG_ELEMS + (size_t)lane * 16;
  const unsigned short* hL  = hfrag + (size_t)lane * 16;
  const unsigned short* rhL = rhfrag + (size_t)lane * 16;

  for (int t = 0; t < S; ++t) {
    const unsigned short* xL = xf + (size_t)t * TSTEP_FRAGS * FRAG_ELEMS + (size_t)lane * 16;
    unsigned short* yf_t = yf + (size_t)t * TSTEP_FRAGS * FRAG_ELEMS;

    // ===== phase 1: zpre/rpre over [x||h] K=2048 ; hx = x @ Wh[:H] =====
    v8f accA0 = {}, accA1 = {}, accB0 = {}, accB1 = {};
    if (isZ) {
      // z gate + hx path: 4 WMMAs per kt, no divergence inside the pipeline
      v16bf a0 = ldf(xL, 0), a1 = ldf(xL, 1);
      v16bf bA = *(const v16bf*)wAL;
      v16bf bB = *(const v16bf*)wBL;
      for (int kt = 0; kt < KTH - 1; ++kt) {
        const v16bf na0 = ldf(xL, (size_t)(kt + 1) * 2);
        const v16bf na1 = ldf(xL, (size_t)(kt + 1) * 2 + 1);
        const v16bf nbA = *(const v16bf*)(wAL + (size_t)(kt + 1) * KS);
        const v16bf nbB = *(const v16bf*)(wBL + (size_t)(kt + 1) * KS);
        const int pkt = (kt + 5 < KTH) ? kt + 5 : kt;
        __builtin_prefetch(wAL + (size_t)pkt * KS, 0, 1);
        accA0 = wmma_bf16(a0, bA, accA0);
        accA1 = wmma_bf16(a1, bA, accA1);
        accB0 = wmma_bf16(a0, bB, accB0);
        accB1 = wmma_bf16(a1, bB, accB1);
        a0 = na0; a1 = na1; bA = nbA; bB = nbB;
      }
      accA0 = wmma_bf16(a0, bA, accA0);
      accA1 = wmma_bf16(a1, bA, accA1);
      accB0 = wmma_bf16(a0, bB, accB0);
      accB1 = wmma_bf16(a1, bB, accB1);
    } else {
      // r gate path: 2 WMMAs per kt
      v16bf a0 = ldf(xL, 0), a1 = ldf(xL, 1);
      v16bf bA = *(const v16bf*)wAL;
      for (int kt = 0; kt < KTH - 1; ++kt) {
        const v16bf na0 = ldf(xL, (size_t)(kt + 1) * 2);
        const v16bf na1 = ldf(xL, (size_t)(kt + 1) * 2 + 1);
        const v16bf nbA = *(const v16bf*)(wAL + (size_t)(kt + 1) * KS);
        const int pkt = (kt + 5 < KTH) ? kt + 5 : kt;
        __builtin_prefetch(wAL + (size_t)pkt * KS, 0, 1);
        accA0 = wmma_bf16(a0, bA, accA0);
        accA1 = wmma_bf16(a1, bA, accA1);
        a0 = na0; a1 = na1; bA = nbA;
      }
      accA0 = wmma_bf16(a0, bA, accA0);
      accA1 = wmma_bf16(a1, bA, accA1);
    }
    {  // h rows of K (k >= 1024): common to z and r paths
      v16bf a0 = ldf(hL, 0), a1 = ldf(hL, 1);
      v16bf bA = *(const v16bf*)(wAL + (size_t)KTH * KS);
      for (int kt = 0; kt < KTH - 1; ++kt) {
        const v16bf na0 = ldf(hL, (size_t)(kt + 1) * 2);
        const v16bf na1 = ldf(hL, (size_t)(kt + 1) * 2 + 1);
        const v16bf nbA = *(const v16bf*)(wAL + (size_t)(KTH + kt + 1) * KS);
        const int pkt = (kt + 5 < KTH) ? kt + 5 : kt;
        __builtin_prefetch(wAL + (size_t)(KTH + pkt) * KS, 0, 1);
        accA0 = wmma_bf16(a0, bA, accA0);
        accA1 = wmma_bf16(a1, bA, accA1);
        a0 = na0; a1 = na1; bA = nbA;
      }
      accA0 = wmma_bf16(a0, bA, accA0);
      accA1 = wmma_bf16(a1, bA, accA1);
    }
    // phase-1 epilogue: C layout m = i + 8*(lane>=16), n = lane&15
    {
      const int n = ntA * 16 + (lane & 15);
      const int mb = (lane >> 4) << 3;
      if (isZ) {
#pragma unroll
        for (int i = 0; i < 8; ++i) {
          const int m0 = i + mb, m1 = 16 + i + mb;
          zbuf[m0 * GH + n] = sigm(accA0[i] + bz[n]);
          zbuf[m1 * GH + n] = sigm(accA1[i] + bz[n]);
          hxbuf[m0 * GH + n] = accB0[i];
          hxbuf[m1 * GH + n] = accB1[i];
        }
      } else {
#pragma unroll
        for (int i = 0; i < 8; ++i) {
          const int m0 = i + mb, m1 = 16 + i + mb;
          const float r0 = sigm(accA0[i] + br[n]);
          const float r1 = sigm(accA1[i] + br[n]);
          store_afrag(rhfrag, m0, n, r0 * h[m0 * GH + n]);
          store_afrag(rhfrag, m1, n, r1 * h[m1 * GH + n]);
        }
      }
    }
    gbarrier(bar, gen, nblocks);

    // ===== phase 2: h~ = tanh(hx + rh @ Wh[H:] + bh); h' = z*h + (1-z)*h~ =====
    if (isZ) {
      v8f accC0 = {}, accC1 = {};
      v16bf a0 = ldf(rhL, 0), a1 = ldf(rhL, 1);
      v16bf bC = *(const v16bf*)wCL;
      for (int kt = 0; kt < KTH - 1; ++kt) {
        const v16bf na0 = ldf(rhL, (size_t)(kt + 1) * 2);
        const v16bf na1 = ldf(rhL, (size_t)(kt + 1) * 2 + 1);
        const v16bf nbC = *(const v16bf*)(wCL + (size_t)(kt + 1) * KS);
        accC0 = wmma_bf16(a0, bC, accC0);
        accC1 = wmma_bf16(a1, bC, accC1);
        a0 = na0; a1 = na1; bC = nbC;
      }
      accC0 = wmma_bf16(a0, bC, accC0);
      accC1 = wmma_bf16(a1, bC, accC1);

      const int n = wgid * 16 + (lane & 15);
      const int mb = (lane >> 4) << 3;
#pragma unroll
      for (int i = 0; i < 8; ++i) {
        const int m0 = i + mb, m1 = 16 + i + mb;
        {
          const float ht = tanhf(accC0[i] + hxbuf[m0 * GH + n] + bh[n]);
          const float z = zbuf[m0 * GH + n];
          const float hn = z * h[m0 * GH + n] + (1.0f - z) * ht;
          h[m0 * GH + n] = hn;
          store_afrag(hfrag, m0, n, hn);
          store_afrag(yf_t, m0, n, hn);
        }
        {
          const float ht = tanhf(accC1[i] + hxbuf[m1 * GH + n] + bh[n]);
          const float z = zbuf[m1 * GH + n];
          const float hn = z * h[m1 * GH + n] + (1.0f - z) * ht;
          h[m1 * GH + n] = hn;
          store_afrag(hfrag, m1, n, hn);
          store_afrag(yf_t, m1, n, hn);
        }
      }
    }
    gbarrier(bar, gen, nblocks);
  }
}

// ---------------- output projection: 2 t-slices per block, pipelined ----------------
// grid: (VTGT/256, STGT/2); 8 waves, each wave: 2t x 2mt x 2nt = 8 accumulators
__global__ __launch_bounds__(256) void proj_gemm(
    const unsigned short* __restrict__ Af, const unsigned short* __restrict__ Bp,
    const float* __restrict__ bias, float* __restrict__ out) {
  const int lane = threadIdx.x & 31;
  const int w = threadIdx.x >> 5;
  const int t0 = blockIdx.y * 2;
  const int nt0 = blockIdx.x * 16 + w * 2;

  const size_t KS = (size_t)NTB * FRAG_ELEMS;  // per-kt stride in packed out_W
  const unsigned short* aL0 = Af + (size_t)t0 * TSTEP_FRAGS * FRAG_ELEMS + (size_t)lane * 16;
  const unsigned short* aL1 = aL0 + (size_t)TSTEP_FRAGS * FRAG_ELEMS;
  const unsigned short* bL = Bp + (size_t)nt0 * FRAG_ELEMS + (size_t)lane * 16;

  v8f c000 = {}, c001 = {}, c010 = {}, c011 = {};  // t0: [mt][nt]
  v8f c100 = {}, c101 = {}, c110 = {}, c111 = {};  // t1

  v16bf x0 = ldf(aL0, 0), x1 = ldf(aL0, 1);
  v16bf y0 = ldf(aL1, 0), y1 = ldf(aL1, 1);
  v16bf b0 = *(const v16bf*)bL;
  v16bf b1 = *(const v16bf*)(bL + FRAG_ELEMS);
  for (int kt = 0; kt < KTH - 1; ++kt) {
    const v16bf nx0 = ldf(aL0, (size_t)(kt + 1) * 2);
    const v16bf nx1 = ldf(aL0, (size_t)(kt + 1) * 2 + 1);
    const v16bf ny0 = ldf(aL1, (size_t)(kt + 1) * 2);
    const v16bf ny1 = ldf(aL1, (size_t)(kt + 1) * 2 + 1);
    const v16bf nb0 = *(const v16bf*)(bL + (size_t)(kt + 1) * KS);
    const v16bf nb1 = *(const v16bf*)(bL + (size_t)(kt + 1) * KS + FRAG_ELEMS);
    const int pkt = (kt + 3 < KTH) ? kt + 3 : kt;
    __builtin_prefetch(bL + (size_t)pkt * KS, 0, 1);
    c000 = wmma_bf16(x0, b0, c000);
    c001 = wmma_bf16(x0, b1, c001);
    c010 = wmma_bf16(x1, b0, c010);
    c011 = wmma_bf16(x1, b1, c011);
    c100 = wmma_bf16(y0, b0, c100);
    c101 = wmma_bf16(y0, b1, c101);
    c110 = wmma_bf16(y1, b0, c110);
    c111 = wmma_bf16(y1, b1, c111);
    x0 = nx0; x1 = nx1; y0 = ny0; y1 = ny1; b0 = nb0; b1 = nb1;
  }
  c000 = wmma_bf16(x0, b0, c000);
  c001 = wmma_bf16(x0, b1, c001);
  c010 = wmma_bf16(x1, b0, c010);
  c011 = wmma_bf16(x1, b1, c011);
  c100 = wmma_bf16(y0, b0, c100);
  c101 = wmma_bf16(y0, b1, c101);
  c110 = wmma_bf16(y1, b0, c110);
  c111 = wmma_bf16(y1, b1, c111);

  const int mb = (lane >> 4) << 3;
  const int nlo = lane & 15;
  const int v0 = (nt0 + 0) * 16 + nlo;
  const int v1 = (nt0 + 1) * 16 + nlo;
  const float bias0 = bias[v0], bias1 = bias[v1];
#pragma unroll
  for (int i = 0; i < 8; ++i) {
    const int b0r = i + mb, b1r = 16 + i + mb;  // batch rows
    out[((size_t)b0r * STGT + t0) * VTGT + v0] = c000[i] + bias0;
    out[((size_t)b0r * STGT + t0) * VTGT + v1] = c001[i] + bias1;
    out[((size_t)b1r * STGT + t0) * VTGT + v0] = c010[i] + bias0;
    out[((size_t)b1r * STGT + t0) * VTGT + v1] = c011[i] + bias1;
    out[((size_t)b0r * STGT + t0 + 1) * VTGT + v0] = c100[i] + bias0;
    out[((size_t)b0r * STGT + t0 + 1) * VTGT + v1] = c101[i] + bias1;
    out[((size_t)b1r * STGT + t0 + 1) * VTGT + v0] = c110[i] + bias0;
    out[((size_t)b1r * STGT + t0 + 1) * VTGT + v1] = c111[i] + bias1;
  }
}

// ---------------- host side ----------------
extern "C" void kernel_launch(void* const* d_in, const int* in_sizes, int n_in,
                              void* d_out, int out_size, void* d_ws, size_t ws_size,
                              hipStream_t stream) {
  const float* src_emb = (const float*)d_in[0];
  const float* tgt_emb = (const float*)d_in[1];
  const float* Wgru[2][3] = {{(const float*)d_in[2], (const float*)d_in[4], (const float*)d_in[6]},
                             {(const float*)d_in[8], (const float*)d_in[10], (const float*)d_in[12]}};
  const float* bgru[2][3] = {{(const float*)d_in[3], (const float*)d_in[5], (const float*)d_in[7]},
                             {(const float*)d_in[9], (const float*)d_in[11], (const float*)d_in[13]}};
  const float* out_W = (const float*)d_in[14];
  const float* out_b = (const float*)d_in[15];
  const int* source = (const int*)d_in[16];
  const int* target = (const int*)d_in[17];
  (void)in_sizes; (void)n_in; (void)out_size; (void)ws_size;

  // workspace layout (256B aligned)
  char* base = (char*)d_ws;
  size_t off = 0;
  auto take = [&](size_t bytes) { size_t r = off; off += (bytes + 255) & ~(size_t)255; return r; };
  unsigned* bar = (unsigned*)(base + take(64 * sizeof(unsigned)));
  float* hbuf = (float*)(base + take((size_t)2 * GB * GH * 4));
  unsigned short* hfrag = (unsigned short*)(base + take((size_t)2 * TSTEP_FRAGS * FRAG_ELEMS * 2));
  unsigned short* rhfrag = (unsigned short*)(base + take((size_t)TSTEP_FRAGS * FRAG_ELEMS * 2));
  float* zb  = (float*)(base + take((size_t)GB * GH * 4));
  float* hxb = (float*)(base + take((size_t)GB * GH * 4));
  unsigned short* Xf = (unsigned short*)(base + take((size_t)SSRC * TSTEP_FRAGS * FRAG_ELEMS * 2));
  unsigned short* Yf = (unsigned short*)(base + take((size_t)SSRC * TSTEP_FRAGS * FRAG_ELEMS * 2));
  const size_t WSLOT = (size_t)GK2 * GH;
  unsigned short* wp  = (unsigned short*)(base + take(12 * WSLOT * 2));        // 48 MB
  unsigned short* owp = (unsigned short*)(base + take((size_t)GH * VTGT * 2)); // 65.5 MB

  init_ws<<<64, 256, 0, stream>>>(bar, hbuf, (unsigned*)hfrag);

  // source embeddings -> Xf fragments
  gather_embed_frag<<<SSRC * 8, 256, 0, stream>>>(src_emb, source, Xf, SSRC);

  // pack weights (stack s: 0=enc,1=dec; gate g: 0=z,1=r,2=h; layer l)
  const int gru_pack_total = (GK2 / 32) * (GH / 16) * 32;  // 131072
  for (int s = 0; s < 2; ++s)
    for (int g = 0; g < 3; ++g)
      for (int l = 0; l < 2; ++l) {
        const int slot = (s * 3 + g) * 2 + l;
        pack_weight_b<<<(gru_pack_total + 255) / 256, 256, 0, stream>>>(
            Wgru[s][g] + (size_t)l * GK2 * GH, wp + (size_t)slot * WSLOT, GK2, GH);
      }
  const int ow_pack_total = (GH / 32) * (VTGT / 16) * 32;  // 2,048,000
  pack_weight_b<<<(ow_pack_total + 255) / 256, 256, 0, stream>>>(out_W, owp, GH, VTGT);

  auto wslot = [&](int s, int g, int l) { return wp + (size_t)((s * 3 + g) * 2 + l) * WSLOT; };
  const size_t HFR = (size_t)TSTEP_FRAGS * FRAG_ELEMS;

  // encoder (h0 = 0 from init_ws)
  gru_layer<<<GRU_BLOCKS, 256, 0, stream>>>(Xf, Yf, SSRC, hbuf, hfrag, rhfrag,
      wslot(0, 0, 0), wslot(0, 1, 0), wslot(0, 2, 0),
      bgru[0][0], bgru[0][1], bgru[0][2], zb, hxb, bar + 0, GRU_BLOCKS);
  gru_layer<<<GRU_BLOCKS, 256, 0, stream>>>(Yf, Xf, SSRC, hbuf + (size_t)GB * GH, hfrag + HFR, rhfrag,
      wslot(0, 0, 1), wslot(0, 1, 1), wslot(0, 2, 1),
      bgru[0][0] + GH, bgru[0][1] + GH, bgru[0][2] + GH, zb, hxb, bar + 1, GRU_BLOCKS);

  // target embeddings -> Yf fragments (encoder outputs discarded; h state persists)
  gather_embed_frag<<<STGT * 8, 256, 0, stream>>>(tgt_emb, target, Yf, STGT);

  // decoder (h0 = encoder finals, continues in hbuf/hfrag)
  gru_layer<<<GRU_BLOCKS, 256, 0, stream>>>(Yf, Xf, STGT, hbuf, hfrag, rhfrag,
      wslot(1, 0, 0), wslot(1, 1, 0), wslot(1, 2, 0),
      bgru[1][0], bgru[1][1], bgru[1][2], zb, hxb, bar + 2, GRU_BLOCKS);
  gru_layer<<<GRU_BLOCKS, 256, 0, stream>>>(Xf, Yf, STGT, hbuf + (size_t)GB * GH, hfrag + HFR, rhfrag,
      wslot(1, 0, 1), wslot(1, 1, 1), wslot(1, 2, 1),
      bgru[1][0] + GH, bgru[1][1] + GH, bgru[1][2] + GH, zb, hxb, bar + 3, GRU_BLOCKS);

  // logits = dec_out @ out_W + out_b (A fragments straight from Yf, 2 t-slices per block)
  proj_gemm<<<dim3(VTGT / 256, STGT / 2), 256, 0, stream>>>(Yf, owp, out_b, (float*)d_out);
}
